// ScaledDotProductAttention_81466939671043
// MI455X (gfx1250) — matrix operand order repaired and checked by reading, not compiled, not run on hardware
//
#include <hip/hip_runtime.h>
#include <hip/hip_bf16.h>

typedef unsigned short u16;
typedef __attribute__((ext_vector_type(8)))  __bf16 v8bf;
typedef __attribute__((ext_vector_type(16))) __bf16 v16bf;
typedef __attribute__((ext_vector_type(8)))  float  v8f;

#define B_DIM 4
#define S_LEN 4096
#define D_DIM 512
#define BR    64                     // query rows per block
#define BC    128                    // kv chunk width
#define PS    (BC + 8)               // padded LDS row stride (stays 16B aligned)
#define SCALE 0.04419417382415922f   // 1/sqrt(512)

__device__ __forceinline__ u16 f32_to_bf16(float f) {
  union { float f; unsigned u; } v; v.f = f;
  unsigned r = v.u + 0x7FFFu + ((v.u >> 16) & 1u);
  return (u16)(r >> 16);
}

// A-fragment (16x32 bf16, MxK): lane L holds row M=L&15.
// halves 0..7  = K[hi*8 .. hi*8+7],  halves 8..15 = K[16+hi*8 .. +7]  (hi = L>>4)
__device__ __forceinline__ v16bf afrag_from_row(const u16* __restrict__ rp, int hi) {
  v8bf lo = *(const v8bf*)(rp + hi * 8);
  v8bf hb = *(const v8bf*)(rp + 16 + hi * 8);
  v16bf a;
#pragma unroll
  for (int i = 0; i < 8; ++i) { a[i] = lo[i]; a[i + 8] = hb[i]; }
  return a;
}

// B-fragment (32x16 bf16, KxN): lane L holds column N=L&15; halves 0..7 = K[hi*8..+7], 8..15 = K[16+hi*8..+7].
__device__ __forceinline__ v16bf bfrag_from_row(const u16* __restrict__ rp, int hi) {
  v8bf lo = *(const v8bf*)(rp + hi * 8);
  v8bf hb = *(const v8bf*)(rp + 16 + hi * 8);
  v16bf b;
#pragma unroll
  for (int i = 0; i < 8; ++i) { b[i] = lo[i]; b[i + 8] = hb[i]; }
  return b;
}

__device__ __forceinline__ v8f wmma_bf16(v16bf a, v16bf b, v8f c) {
  return __builtin_amdgcn_wmma_f32_16x16x32_bf16(false, a, false, b, (short)0, c, false, false);
}

// ---------------- Kernel 1: fp32 -> bf16 conversion ----------------
__global__ void cvt_kernel(const float* __restrict__ x,
                           const float* __restrict__ wq,
                           const float* __restrict__ wk,
                           const float* __restrict__ wv,
                           u16* __restrict__ xb, u16* __restrict__ wqb,
                           u16* __restrict__ wkb, u16* __restrict__ wvb,
                           int nx, int nw) {
  int total = nx + 3 * nw;
  for (int i = blockIdx.x * blockDim.x + threadIdx.x; i < total;
       i += gridDim.x * blockDim.x) {
    if (i < nx) {
      xb[i] = f32_to_bf16(x[i]);
    } else {
      int j = i - nx;
      if (j < nw)          wqb[j]          = f32_to_bf16(wq[j]);
      else if (j < 2 * nw) wkb[j - nw]     = f32_to_bf16(wk[j - nw]);
      else                 wvb[j - 2 * nw] = f32_to_bf16(wv[j - 2 * nw]);
    }
  }
}

// ---------------- Kernel 2: projections Q,K (row-major) and V^T ----------------
// out[m,n] = sum_k x[m,k] * W[n,k]   (torch Linear: x @ W^T)
__global__ __launch_bounds__(256) void proj_kernel(
    const u16* __restrict__ xb, const u16* __restrict__ wq,
    const u16* __restrict__ wk, const u16* __restrict__ wv,
    u16* __restrict__ Qb, u16* __restrict__ Kb, u16* __restrict__ Vt) {
  const int tid  = threadIdx.x;
  const int lane = tid & 31;
  const int wave = tid >> 5;
  const int li   = lane & 15;
  const int hi   = lane >> 4;

  const int m0 = blockIdx.x * 16;                 // row tile in flattened [B*S]
  const int n0 = (blockIdx.y * 8 + wave) * 16;    // col tile in [D]
  const int z  = blockIdx.z;                      // 0=Q 1=K 2=V
  const u16* W = (z == 0) ? wq : ((z == 1) ? wk : wv);

  const u16* arow = xb + (size_t)(m0 + li) * D_DIM;
  const u16* brow = W  + (size_t)(n0 + li) * D_DIM;

  v8f acc = {};
#pragma unroll
  for (int kc = 0; kc < 16; ++kc) {
    v16bf a  = afrag_from_row(arow + kc * 32, hi);
    v16bf bf = bfrag_from_row(brow + kc * 32, hi);
    acc = wmma_bf16(a, bf, acc);
  }

  if (z < 2) {
    u16* outp = (z == 0) ? Qb : Kb;
#pragma unroll
    for (int j = 0; j < 8; ++j) {
      int m = m0 + j + hi * 8;
      outp[(size_t)m * D_DIM + n0 + li] = f32_to_bf16(acc[j]);
    }
  } else {
    // store V transposed: Vt[b, d, s]
    int bidx  = m0 >> 12;      // /S
    int sbase = m0 & (S_LEN - 1);
#pragma unroll
    for (int j = 0; j < 8; ++j) {
      int c = n0 + li;
      int s = sbase + j + hi * 8;
      Vt[((size_t)bidx * D_DIM + c) * S_LEN + s] = f32_to_bf16(acc[j]);
    }
  }
}

// ---------------- Kernel 3: causal flash attention, Br=64 ----------------
// Block: 256 threads = 8 waves, one 64-row query tile per block (Q staged in LDS).
// Per kv chunk of 128 columns:
//   wave w computes 4 score subtiles (rows rt*16..) for cols [c0+16w, +16)
//     -> each K B-fragment loaded once, used by 4 WMMAs (64 WMMAs/wave)
//   cooperative online softmax over the 64x128 chunk (4 threads per row)
//   wave w accumulates its 64 owned output columns for all 64 rows:
//     o[4][4] C-tiles, each V B-fragment loaded once, used 4x (64 WMMAs/wave)
__global__ __launch_bounds__(256) void attn_kernel(
    const u16* __restrict__ Qb, const u16* __restrict__ Kb,
    const u16* __restrict__ Vt, float* __restrict__ out) {
  __shared__ __align__(16) u16   Qld[BR * D_DIM];   // 64 KB
  __shared__ __align__(16) float Sld[BR * PS];      // ~34.8 KB (padded)
  __shared__ __align__(16) u16   Pld[BR * PS];      // ~17.4 KB (padded)
  __shared__ float mrow[BR], lrow[BR], fac[BR];

  const int tid  = threadIdx.x;
  const int lane = tid & 31;
  const int wave = tid >> 5;
  const int li   = lane & 15;
  const int hi   = lane >> 4;

  const int q0 = blockIdx.x * BR;
  const int b  = blockIdx.y;

  const u16* Kbase = Kb + (size_t)b * S_LEN * D_DIM;
  const u16* Vbase = Vt + (size_t)b * D_DIM * S_LEN;

  // ---- stage Q tile (64 x 512 bf16, rows contiguous) into LDS ----
  {
    const uint4* src =
        (const uint4*)(Qb + ((size_t)b * S_LEN + q0) * D_DIM);
    uint4* dst = (uint4*)Qld;
    const int nchunks = BR * D_DIM / 8;  // 8 u16 per uint4
#pragma unroll 4
    for (int i = tid; i < nchunks; i += 256) dst[i] = src[i];
  }
  if (tid < BR) { mrow[tid] = -3.0e38f; lrow[tid] = 0.0f; }

  v8f o[4][4];  // [row-subtile][col-tile], 64 rows x 64 owned cols, f32
#pragma unroll
  for (int rt = 0; rt < 4; ++rt)
#pragma unroll
    for (int nt = 0; nt < 4; ++nt) o[rt][nt] = (v8f){};

  __syncthreads();

  const int nsteps = (q0 + BR + BC - 1) / BC;
  for (int t = 0; t < nsteps; ++t) {
    const int c0 = t * BC;

    // ---- scores: S = Q K^T * scale, causal-masked ----
    {
      const int n = c0 + wave * 16 + li;
      const u16* krow = Kbase + (size_t)n * D_DIM;
      v8f acc[4];
#pragma unroll
      for (int rt = 0; rt < 4; ++rt) acc[rt] = (v8f){};
#pragma unroll
      for (int kc = 0; kc < 16; ++kc) {
        v16bf bf = bfrag_from_row(krow + kc * 32, hi);
#pragma unroll
        for (int rt = 0; rt < 4; ++rt) {
          v16bf a =
              afrag_from_row(&Qld[(rt * 16 + li) * D_DIM + kc * 32], hi);
          acc[rt] = wmma_bf16(a, bf, acc[rt]);
        }
      }
#pragma unroll
      for (int rt = 0; rt < 4; ++rt) {
#pragma unroll
        for (int j = 0; j < 8; ++j) {
          int m = rt * 16 + j + hi * 8;
          float v = acc[rt][j] * SCALE;
          if (n > q0 + m) v = -3.0e38f;  // causal mask
          Sld[m * PS + wave * 16 + li] = v;
        }
      }
    }
    __syncthreads();

    // ---- online softmax over the 64x128 chunk (4 threads per row) ----
    {
      const int row = tid >> 2;   // 0..63
      const int sub = tid & 3;    // 32 cols each
      const float* srow = &Sld[row * PS + sub * 32];
      float mx = -3.0e38f;
#pragma unroll
      for (int j = 0; j < 32; ++j) mx = fmaxf(mx, srow[j]);
      mx = fmaxf(mx, __shfl_xor(mx, 1, 4));
      mx = fmaxf(mx, __shfl_xor(mx, 2, 4));
      float mold = mrow[row];
      float mnew = fmaxf(mold, mx);
      float sum = 0.0f;
      u16* prow = &Pld[row * PS + sub * 32];
#pragma unroll
      for (int j = 0; j < 32; ++j) {
        float p = __expf(srow[j] - mnew);
        sum += p;
        prow[j] = f32_to_bf16(p);
      }
      sum += __shfl_xor(sum, 1, 4);
      sum += __shfl_xor(sum, 2, 4);
      if (sub == 0) {
        float f = __expf(mold - mnew);
        fac[row]  = f;
        lrow[row] = lrow[row] * f + sum;
        mrow[row] = mnew;
      }
    }
    __syncthreads();

    // ---- O update: rescale then accumulate P x V for owned 64 columns ----
    {
#pragma unroll
      for (int rt = 0; rt < 4; ++rt) {
#pragma unroll
        for (int j = 0; j < 8; ++j) {
          float f = fac[rt * 16 + j + hi * 8];
#pragma unroll
          for (int nt = 0; nt < 4; ++nt) o[rt][nt][j] *= f;
        }
      }
      const int n0 = wave * 64;
#pragma unroll
      for (int kc = 0; kc < 4; ++kc) {
        v16bf vb[4];
#pragma unroll
        for (int nt = 0; nt < 4; ++nt) {
          const u16* vrow =
              Vbase + (size_t)(n0 + nt * 16 + li) * S_LEN + c0 + kc * 32;
          vb[nt] = bfrag_from_row(vrow, hi);
        }
#pragma unroll
        for (int rt = 0; rt < 4; ++rt) {
          v16bf pa = afrag_from_row(&Pld[(rt * 16 + li) * PS + kc * 32], hi);
#pragma unroll
          for (int nt = 0; nt < 4; ++nt)
            o[rt][nt] = wmma_bf16(pa, vb[nt], o[rt][nt]);
        }
      }
    }
    // next iteration: score phase only writes Sld (fully consumed by softmax
    // before the post-score barrier); Pld/fac are rewritten only after that
    // barrier, by which time every wave finished this O update -> safe.
  }

  // ---- normalize by l and store fp32 output ----
  const int n0 = wave * 64;
#pragma unroll
  for (int rt = 0; rt < 4; ++rt) {
#pragma unroll
    for (int j = 0; j < 8; ++j) {
      int m = rt * 16 + j + hi * 8;
      float linv = 1.0f / lrow[m];
#pragma unroll
      for (int nt = 0; nt < 4; ++nt) {
        out[((size_t)b * S_LEN + q0 + m) * D_DIM + n0 + nt * 16 + li] =
            o[rt][nt][j] * linv;
      }
    }
  }
}

// ---------------- host launcher ----------------
extern "C" void kernel_launch(void* const* d_in, const int* in_sizes, int n_in,
                              void* d_out, int out_size, void* d_ws, size_t ws_size,
                              hipStream_t stream) {
  (void)in_sizes; (void)n_in; (void)out_size; (void)ws_size;
  const float* x  = (const float*)d_in[0];
  const float* wq = (const float*)d_in[1];
  const float* wk = (const float*)d_in[2];
  const float* wv = (const float*)d_in[3];

  const size_t nx = (size_t)B_DIM * S_LEN * D_DIM; // 8M elems
  const size_t nw = (size_t)D_DIM * D_DIM;         // 256K elems

  u16* p   = (u16*)d_ws;
  u16* xb  = p; p += nx;
  u16* wqb = p; p += nw;
  u16* wkb = p; p += nw;
  u16* wvb = p; p += nw;
  u16* Qb  = p; p += nx;
  u16* Kb  = p; p += nx;
  u16* Vt  = p; p += nx;

  cvt_kernel<<<4096, 256, 0, stream>>>(x, wq, wk, wv, xb, wqb, wkb, wvb,
                                       (int)nx, (int)nw);

  // GEMM: M = B*S = 16384 (1024 row tiles), N = 512 (32 col tiles / 8 waves), z = Q/K/V
  proj_kernel<<<dim3(B_DIM * S_LEN / 16, 4, 3), 256, 0, stream>>>(
      xb, wqb, wkb, wvb, Qb, Kb, Vt);

  // Attention: one block per 64-row query tile per batch
  attn_kernel<<<dim3(S_LEN / BR, B_DIM), 256, 0, stream>>>(Qb, Kb, Vt,
                                                           (float*)d_out);
}
